// LanguageIDModel_85813446574656
// MI455X (gfx1250) — compile-verified, matrix operand order checked by reading, not run on hardware
//
#include <hip/hip_runtime.h>
#include <cstdint>
#include <cstddef>

typedef __bf16 v16bf __attribute__((ext_vector_type(16)));
typedef float  v8f   __attribute__((ext_vector_type(8)));

#define NUM_CHARS 47
#define HIDDEN    128
#define G3        384
#define SEQ_L     256
#define BATCH     4096

__device__ __forceinline__ unsigned short f2bf(float f) {
    __bf16 h = (__bf16)f;                 // native RNE convert on gfx1250
    unsigned short s;
    __builtin_memcpy(&s, &h, 2);
    return s;
}
__device__ __forceinline__ float bf2f(unsigned short s) {
    return __uint_as_float(((unsigned int)s) << 16);
}
__device__ __forceinline__ float sig_f(float x)  { return 1.0f / (1.0f + __expf(-x)); }
__device__ __forceinline__ float tanh_f(float x) { return 1.0f - 2.0f / (__expf(2.0f * x) + 1.0f); }

union FragB   { uint4 q[2]; v16bf v; };
union FragAcc { v8f v; float f[8]; };

// ---------------------------------------------------------------------------
// Kernel 1: fold embedding+input-proj weights; pre-swizzle WMMA fragments.
//   W_fold[g][c] = sum_h W_ih[g][h]*W_emb[h][c]   (bf16, A-fragment layout)
//   b_fold[g]    = b_ih[g] + W_ih[g,:]@b_emb + (g<256 ? b_hh[g] : 0)
//   whh[k][n]    = W_hh[n][k]                      (bf16, B-fragment layout)
// Fragment element i (0..15), lane half hi: k_off = hi*8 + (i/8)*16 + i%8
// ---------------------------------------------------------------------------
__global__ void __launch_bounds__(256) fold_kernel(
    const float* __restrict__ W_emb, const float* __restrict__ b_emb,
    const float* __restrict__ W_ih,  const float* __restrict__ b_ih,
    const float* __restrict__ W_hh,  const float* __restrict__ b_hh,
    unsigned short* __restrict__ wf, float* __restrict__ bfold,
    unsigned short* __restrict__ whh)
{
    const int tid = threadIdx.x;

    // W_fold fragments: [gt(24)][kt(2)][lane(32)][i(16)]
    for (int idx = tid; idx < 24 * 2 * 32 * 16; idx += 256) {
        int i    = idx & 15;
        int lane = (idx >> 4) & 31;
        int kt   = (idx >> 9) & 1;
        int gt   = idx >> 10;
        int hi = lane >> 4, m = lane & 15;
        int koff = (hi ? 8 : 0) + ((i >> 3) ? 16 : 0) + (i & 7);
        int c = kt * 32 + koff;
        int g = gt * 16 + m;
        float v = 0.0f;
        if (c < NUM_CHARS) {
            for (int h = 0; h < HIDDEN; ++h)
                v += W_ih[g * HIDDEN + h] * W_emb[h * NUM_CHARS + c];
        }
        wf[idx] = f2bf(v);
    }

    // b_fold (f32); fold b_hh for r,z gates (their bias is purely additive pre-activation)
    for (int g = tid; g < G3; g += 256) {
        float v = b_ih[g];
        for (int h = 0; h < HIDDEN; ++h) v += W_ih[g * HIDDEN + h] * b_emb[h];
        if (g < 2 * HIDDEN) v += b_hh[g];
        bfold[g] = v;
    }

    // W_hh B-fragments: [kt(4)][nt(24)][lane(32)][i(16)],  B[k][n] = W_hh[n][k]
    for (int idx = tid; idx < 4 * 24 * 32 * 16; idx += 256) {
        int i    = idx & 15;
        int lane = (idx >> 4) & 31;
        int rest = idx >> 9;
        int nt   = rest % 24;
        int kt   = rest / 24;
        int hi = lane >> 4, n = lane & 15;
        int koff = (hi ? 8 : 0) + ((i >> 3) ? 16 : 0) + (i & 7);
        int k = kt * 32 + koff;
        int g = nt * 16 + n;
        whh[idx] = f2bf(W_hh[g * HIDDEN + k]);
    }
}

// ---------------------------------------------------------------------------
// Kernel 2: x_projT[l][g][b] (bf16) = (xs[l] @ W_fold^T + b_fold) transposed.
// Grid: (B/128, L). 8 waves, each does 24 of the 24x8 (g-tile x b-tile) tiles.
// xs tile staged in LDS as bf16 padded to K=64 (row stride 72 vs bank conflicts).
// ---------------------------------------------------------------------------
__global__ void __launch_bounds__(256) xproj_kernel(
    const float* __restrict__ xs,
    const unsigned short* __restrict__ wf,
    const float* __restrict__ bfold,
    unsigned short* __restrict__ xproj)
{
    __shared__ unsigned short xsl[128 * 72];

    const int l   = blockIdx.y;
    const int b0  = blockIdx.x * 128;
    const int tid = threadIdx.x;

    const float* xrow = xs + ((size_t)l * BATCH + b0) * NUM_CHARS;
    for (int idx = tid; idx < 128 * NUM_CHARS; idx += 256) {
        int bb = idx / NUM_CHARS, c = idx % NUM_CHARS;
        xsl[bb * 72 + c] = f2bf(xrow[(size_t)bb * NUM_CHARS + c]);
    }
    for (int idx = tid; idx < 128 * (64 - NUM_CHARS); idx += 256) {
        int bb = idx / (64 - NUM_CHARS);
        int c  = NUM_CHARS + idx % (64 - NUM_CHARS);
        xsl[bb * 72 + c] = 0;
    }
    __syncthreads();

    const int w = tid >> 5, lane = tid & 31;
    const int hi = lane >> 4, n16 = lane & 15;

    for (int T = w; T < 192; T += 8) {
        int gt = T >> 3, bt = T & 7;
        FragAcc acc;
        acc.v = (v8f){0.f, 0.f, 0.f, 0.f, 0.f, 0.f, 0.f, 0.f};
        for (int kt = 0; kt < 2; ++kt) {
            const v16bf a = *(const v16bf*)&wf[(((gt << 1) | kt) * 32 + lane) << 4];
            FragB bfr;
            const unsigned short* src = &xsl[(bt * 16 + n16) * 72 + kt * 32 + hi * 8];
            bfr.q[0] = *(const uint4*)src;
            bfr.q[1] = *(const uint4*)(src + 16);
            acc.v = __builtin_amdgcn_wmma_f32_16x16x32_bf16(
                false, a, false, bfr.v, (short)0, acc.v, false, false);
        }
        int b = b0 + bt * 16 + n16;
        for (int r = 0; r < 8; ++r) {
            int g = gt * 16 + r + hi * 8;
            float vv = acc.f[r] + bfold[g];
            xproj[((size_t)l * G3 + g) * BATCH + b] = f2bf(vv);
        }
    }
}

// ---------------------------------------------------------------------------
// Kernel 3: persistent GRU recurrence. 128 WGs x 32 batch rows, 256 steps.
// W_hh fragments brought into LDS with CDNA5 async copies (ASYNCcnt-tracked,
// VGPR-bypassing); h double-buffered in LDS (bf16, row stride 136) for the
// A-operand exchange; h also kept in f32 registers.
// Wave w: M-tile = w&1, hidden-col tiles {j0, j0+16} with j0 = (w>>1)*32;
// owns all 3 gates for those columns -> gate math is wave-local.
// ---------------------------------------------------------------------------
__global__ void __launch_bounds__(256) gru_kernel(
    const unsigned short* __restrict__ xproj,
    const unsigned short* __restrict__ whh_g,
    const float* __restrict__ b_hh,
    float* __restrict__ h_last)
{
    __shared__ unsigned short whh_lds[4 * 24 * 32 * 16];  // 98304 B
    __shared__ unsigned short hbuf[2][32 * 136];          // 17408 B

    const int tid = threadIdx.x;
    const int b0  = blockIdx.x * 32;

    {   // W_hh fragments global -> LDS via async DMA (no VGPR staging).
        // Each lane moves 16 B per instruction; 24 instructions per wave.
        for (int i = tid; i < (4 * 24 * 32 * 16) / 8; i += 256) {
            const void* gsrc = (const void*)(whh_g + (size_t)i * 8);
            unsigned lds_off = (unsigned)(uintptr_t)(&whh_lds[(size_t)i * 8]);
            asm volatile("global_load_async_to_lds_b128 %0, %1, off"
                         :: "v"(lds_off), "v"(gsrc) : "memory");
        }
    }
    {   // h0 = 0
        unsigned int* hz = (unsigned int*)hbuf[0];
        for (int i = tid; i < (32 * 136) / 2; i += 256) hz[i] = 0u;
    }
    asm volatile("s_wait_asynccnt 0" ::: "memory");
    __syncthreads();

    const int w = tid >> 5, lane = tid & 31;
    const int hi = lane >> 4, n16 = lane & 15;
    const int m0 = (w & 1) * 16;
    const int jp = w >> 1;          // 0..3
    const int j0 = jp * 32;

    float bn[2];                    // b_hh for the n gate (cannot be folded: scaled by r)
    bn[0] = b_hh[2 * HIDDEN + j0 + n16];
    bn[1] = b_hh[2 * HIDDEN + j0 + 16 + n16];

    float hreg[2][8];
    for (int t = 0; t < 2; ++t)
        for (int r = 0; r < 8; ++r) hreg[t][r] = 0.0f;

    const int    rowoff = (m0 + n16) * 136;          // A-fragment row base
    const size_t xcol   = (size_t)b0 + m0 + hi * 8;  // contiguous 8 batch elems

    // Per-(gate,tile) x_proj base pointers for this lane.
    const unsigned short* xptr[3][2];
    for (int G = 0; G < 3; ++G)
        for (int t = 0; t < 2; ++t) {
            int g = G * HIDDEN + j0 + t * 16 + n16;
            xptr[G][t] = xproj + (size_t)g * BATCH + xcol;
        }
    const size_t xstep = (size_t)G3 * BATCH;         // advance one timestep

    int cur = 0;
    for (int l = 0; l < SEQ_L; ++l) {
        // Prefetch next step's x_proj tiles (h-independent -> fully overlappable).
        if (l + 1 < SEQ_L) {
            for (int G = 0; G < 3; ++G)
                for (int t = 0; t < 2; ++t)
                    __builtin_prefetch(xptr[G][t] + (size_t)(l + 1) * xstep, 0, 0);
        }

        FragAcc acc[3][2];
        for (int G = 0; G < 3; ++G)
            for (int t = 0; t < 2; ++t)
                acc[G][t].v = (v8f){0.f, 0.f, 0.f, 0.f, 0.f, 0.f, 0.f, 0.f};

        const unsigned short* hc = hbuf[cur];
        for (int kt = 0; kt < 4; ++kt) {
            FragB af;
            const unsigned short* ap = &hc[rowoff + kt * 32 + hi * 8];
            af.q[0] = *(const uint4*)ap;
            af.q[1] = *(const uint4*)(ap + 16);
            for (int G = 0; G < 3; ++G) {
                for (int t = 0; t < 2; ++t) {
                    int nt = G * 8 + jp * 2 + t;
                    const v16bf bfr = *(const v16bf*)&whh_lds[((kt * 24 + nt) * 32 + lane) << 4];
                    acc[G][t].v = __builtin_amdgcn_wmma_f32_16x16x32_bf16(
                        false, af.v, false, bfr, (short)0, acc[G][t].v, false, false);
                }
            }
        }

        // x_proj tile values: one b128 per (gate, hidden-tile)
        float xv[3][2][8];
        for (int G = 0; G < 3; ++G) {
            for (int t = 0; t < 2; ++t) {
                uint4 q = *(const uint4*)(xptr[G][t] + (size_t)l * xstep);
                const unsigned int* d = (const unsigned int*)&q;
                for (int p = 0; p < 4; ++p) {
                    xv[G][t][2 * p]     = bf2f((unsigned short)(d[p] & 0xFFFFu));
                    xv[G][t][2 * p + 1] = bf2f((unsigned short)(d[p] >> 16));
                }
            }
        }

        unsigned short* hn = hbuf[cur ^ 1];
        for (int t = 0; t < 2; ++t) {
            for (int r = 0; r < 8; ++r) {
                float rg = sig_f(xv[0][t][r] + acc[0][t].f[r]);
                float zg = sig_f(xv[1][t][r] + acc[1][t].f[r]);
                float ng = tanh_f(xv[2][t][r] + rg * (acc[2][t].f[r] + bn[t]));
                float hv = (1.0f - zg) * ng + zg * hreg[t][r];
                hreg[t][r] = hv;
                hn[(m0 + r + hi * 8) * 136 + j0 + t * 16 + n16] = f2bf(hv);
            }
        }
        __syncthreads();
        cur ^= 1;
    }

    for (int t = 0; t < 2; ++t)
        for (int r = 0; r < 8; ++r)
            h_last[(size_t)(b0 + m0 + r + hi * 8) * HIDDEN + j0 + t * 16 + n16] = hreg[t][r];
}

// ---------------------------------------------------------------------------
// Kernel 4: logits = h_last @ W_out^T + b_out  (4096 x 5, trivial)
// ---------------------------------------------------------------------------
__global__ void __launch_bounds__(256) out_kernel(
    const float* __restrict__ h_last,
    const float* __restrict__ W_out, const float* __restrict__ b_out,
    float* __restrict__ out)
{
    int idx = blockIdx.x * 256 + threadIdx.x;
    if (idx >= BATCH * 5) return;
    int b = idx / 5, o = idx % 5;
    float acc = b_out[o];
    const float* hr = h_last + (size_t)b * HIDDEN;
    const float* wr = W_out + (size_t)o * HIDDEN;
    for (int j = 0; j < HIDDEN; ++j) acc += hr[j] * wr[j];
    out[idx] = acc;
}

// ---------------------------------------------------------------------------
extern "C" void kernel_launch(void* const* d_in, const int* in_sizes, int n_in,
                              void* d_out, int out_size, void* d_ws, size_t ws_size,
                              hipStream_t stream)
{
    const float* xs    = (const float*)d_in[0];
    const float* W_emb = (const float*)d_in[1];
    const float* b_emb = (const float*)d_in[2];
    const float* W_ih  = (const float*)d_in[3];
    const float* b_ih  = (const float*)d_in[4];
    const float* W_hh  = (const float*)d_in[5];
    const float* b_hh  = (const float*)d_in[6];
    const float* W_out = (const float*)d_in[7];
    const float* b_out = (const float*)d_in[8];

    char* ws = (char*)d_ws;
    const size_t XPROJ_BYTES = (size_t)SEQ_L * G3 * BATCH * 2;   // 805,306,368
    unsigned short* xproj = (unsigned short*)ws;
    unsigned short* wf    = (unsigned short*)(ws + XPROJ_BYTES);
    unsigned short* whh   = (unsigned short*)(ws + XPROJ_BYTES + 49152);
    float* bfold          = (float*)(ws + XPROJ_BYTES + 49152 + 98304);
    float* hlast          = (float*)(ws + XPROJ_BYTES + 49152 + 98304 + 1536);

    hipLaunchKernelGGL(fold_kernel, dim3(1), dim3(256), 0, stream,
                       W_emb, b_emb, W_ih, b_ih, W_hh, b_hh, wf, bfold, whh);
    hipLaunchKernelGGL(xproj_kernel, dim3(BATCH / 128, SEQ_L), dim3(256), 0, stream,
                       xs, wf, bfold, xproj);
    hipLaunchKernelGGL(gru_kernel, dim3(BATCH / 32), dim3(256), 0, stream,
                       xproj, whh, b_hh, hlast);
    hipLaunchKernelGGL(out_kernel, dim3((BATCH * 5 + 255) / 256), dim3(256), 0, stream,
                       hlast, W_out, b_out, (float*)d_out);
}
